// GlobalNMTAttentionRNNModel_90537910599927
// MI455X (gfx1250) — matrix operand order repaired and checked by reading, not compiled, not run on hardware
//
#include <hip/hip_runtime.h>
#include <hip/hip_bf16.h>

// ---------------------------------------------------------------------------
// Problem constants (from reference)
// ---------------------------------------------------------------------------
#define B_    256     // batch
#define H_    512     // hidden
#define G_    2048    // 4*H (gates)
#define IN_   300     // input features
#define KP_   320     // IN padded to multiple of 32
#define TLEN_ 40
#define BLEN_ 200
#define TTOT_ 240
#define CH_   512
#define NC_   4
#define NCH_  ((H_ / 32) + (KP_ / 32))   // 16 + 10 = 26 K-chunks per LSTM step

typedef __attribute__((ext_vector_type(8)))  float          floatx8;
typedef __attribute__((ext_vector_type(16))) __bf16         bf16x16;
typedef __attribute__((ext_vector_type(16))) unsigned short ushortx16;
typedef int v2i __attribute__((vector_size(2 * sizeof(int))));
typedef int v4i __attribute__((vector_size(4 * sizeof(int))));

// ---------------------------------------------------------------------------
// CDNA5 async global->LDS staging (ASYNCcnt), with synchronous fallback
// ---------------------------------------------------------------------------
#if defined(__has_builtin)
#if __has_builtin(__builtin_amdgcn_global_load_async_to_lds_b64) && \
    __has_builtin(__builtin_amdgcn_global_load_async_to_lds_b128)
#define USE_ASYNC_LDS 1
#endif
#endif
#ifndef USE_ASYNC_LDS
#define USE_ASYNC_LDS 0
#endif

#define GLOBAL_AS __attribute__((address_space(1)))
#define LDS_AS    __attribute__((address_space(3)))

#if USE_ASYNC_LDS
#define ASYNC_WAIT() asm volatile("s_wait_asynccnt 0" ::: "memory")
#else
#define ASYNC_WAIT() do {} while (0)
#endif

__device__ __forceinline__ void stage_b64(const unsigned short* g,
                                          unsigned short* l) {
#if USE_ASYNC_LDS
  __builtin_amdgcn_global_load_async_to_lds_b64((GLOBAL_AS v2i*)g,
                                                (LDS_AS v2i*)l, 0, 0);
#else
  *(uint2*)l = *(const uint2*)g;
#endif
}

__device__ __forceinline__ void stage_b128(const unsigned short* g,
                                           unsigned short* l) {
#if USE_ASYNC_LDS
  __builtin_amdgcn_global_load_async_to_lds_b128((GLOBAL_AS v4i*)g,
                                                 (LDS_AS v4i*)l, 0, 0);
#else
  *(uint4*)l = *(const uint4*)g;
#endif
}

// ---------------------------------------------------------------------------
// Helpers
// ---------------------------------------------------------------------------
__device__ __forceinline__ unsigned short f2bf(float f) {
  unsigned int u = __float_as_uint(f);
  unsigned int r = u + 0x7FFFu + ((u >> 16) & 1u);   // round-to-nearest-even
  return (unsigned short)(r >> 16);
}

__device__ __forceinline__ float sigmoidf_(float x) {
  return 1.0f / (1.0f + __expf(-x));
}

// A fragment (16x32 bf16, MxK). ISA 7.12.2: lane holds M=lane%16;
// K = (v>=4?16:0) + (lane>=16?8:0) + 2*(v%4) + half
__device__ __forceinline__ bf16x16 frag_a(const unsigned short* sa, int m0,
                                          int lane, int stride) {
  const unsigned short* row = sa + (m0 + (lane & 15)) * stride;
  int kg = (lane >> 4) << 3;          // 0 or 8
  ushortx16 u;
#pragma unroll
  for (int v = 0; v < 8; ++v) {
    int k = ((v & 4) << 2) + kg + ((v & 3) << 1);
    u[2 * v]     = row[k];
    u[2 * v + 1] = row[k + 1];
  }
  return __builtin_bit_cast(bf16x16, u);
}

// B fragment (32x16 bf16, KxN), stored in LDS as W rows [n][k] (B[k][n]=W[n][k]).
// lane holds N=lane%16; K = (v>=4?16:0) + (lane>=16?8:0) + 2*(v%4) + half
__device__ __forceinline__ bf16x16 frag_b(const unsigned short* sw, int n0,
                                          int lane, int stride) {
  const unsigned short* row = sw + (n0 + (lane & 15)) * stride;
  int kg = (lane >> 4) << 3;          // 0 or 8
  ushortx16 u;
#pragma unroll
  for (int v = 0; v < 8; ++v) {
    int k = ((v & 4) << 2) + kg + ((v & 3) << 1);
    u[2 * v]     = row[k];
    u[2 * v + 1] = row[k + 1];
  }
  return __builtin_bit_cast(bf16x16, u);
}

__device__ __forceinline__ floatx8 wmma_bf16(bf16x16 a, bf16x16 b, floatx8 c) {
  return __builtin_amdgcn_wmma_f32_16x16x32_bf16(false, a, false, b,
                                                 (short)0, c, false, false);
}

// ---------------------------------------------------------------------------
// Fused LSTM step: g = h@Whh^T + x_t@Wih^T + bih + bhh; gates; update h,c.
// h state kept in bf16 so ALL staging is a raw byte copy (async-DMA friendly).
// Block: 256 thr = 8 waves = (4 gates) x (2 row-halves). Tile: 32 rows x 64 ch.
// Unified 26-chunk K loop (16 chunks of Whh/h + 10 chunks of Wih/x_t) with
// double-buffered LDS: stage chunk c+1 overlaps the WMMAs of chunk c.
// Grid: (G/4/64=8, B/32=8)
// ---------------------------------------------------------------------------
__global__ __launch_bounds__(256) void lstm_step_kernel(
    const unsigned short* __restrict__ xb_t,   // B_ x KP_ bf16 (this timestep)
    const unsigned short* __restrict__ Wih,    // G_ x KP_ bf16
    const unsigned short* __restrict__ Whh,    // G_ x H_  bf16
    const float* __restrict__ bih, const float* __restrict__ bhh,   // G_
    const unsigned short* __restrict__ h_in,   // B_ x H_ bf16
    const float* __restrict__ c_in,            // B_ x H_ f32
    unsigned short* __restrict__ h_out,        // B_ x H_ bf16
    float* __restrict__ c_out,                 // B_ x H_ f32
    float* __restrict__ hs_out, int hs_stride) // optional f32 history slice
{
  __shared__ unsigned short sa[2][32 * 40];      // A chunk  (32 rows x 32 k)
  __shared__ unsigned short sw[2][4 * 64 * 40];  // W chunk  (4 gates x 64 ch x 32 k)
  __shared__ float          sg[4][32][65];       // gate pre-activations

  const int tid  = threadIdx.x;
  const int lane = tid & 31;
  const int wave = tid >> 5;
  const int wm   = wave & 1;        // row half (0/1)
  const int gq   = wave >> 1;       // gate (0..3)
  const int mblk = blockIdx.y * 32; // batch rows
  const int jblk = blockIdx.x * 64; // hidden channels

  const int arow = tid >> 3, akc = (tid & 7) << 2;   // A staging coords (b64)

  auto stage = [&](int c, int buf) {
    const unsigned short* A;
    const unsigned short* W;
    int ld, k0;
    if (c < H_ / 32) { A = h_in; W = Whh; ld = H_;  k0 = c * 32; }
    else             { A = xb_t; W = Wih; ld = KP_; k0 = (c - H_ / 32) * 32; }
    // A tile: 32 rows x 32 k  (8B per thread)
    stage_b64(A + (size_t)(mblk + arow) * ld + k0 + akc,
              &sa[buf][arow * 40 + akc]);
    // W tile: 4 gates x 64 ch x 32 k  (4 x 16B per thread)
#pragma unroll
    for (int i = 0; i < 4; ++i) {
      int idx = i * 256 + tid;
      int r = idx >> 2, kc = (idx & 3) << 3;
      int g = r >> 6, ch = r & 63;
      stage_b128(W + (size_t)(g * H_ + jblk + ch) * ld + k0 + kc,
                 &sw[buf][r * 40 + kc]);
    }
  };

  floatx8 acc[4];
#pragma unroll
  for (int f = 0; f < 4; ++f)
#pragma unroll
    for (int r = 0; r < 8; ++r) acc[f][r] = 0.0f;

  stage(0, 0);
  for (int c = 0; c < NCH_; ++c) {
    ASYNC_WAIT();            // drain staging of buffer c&1
    __syncthreads();
    if (c + 1 < NCH_) stage(c + 1, (c + 1) & 1);   // overlaps WMMAs below
    bf16x16 a = frag_a(sa[c & 1], wm * 16, lane, 40);
#pragma unroll
    for (int f = 0; f < 4; ++f) {
      bf16x16 b = frag_b(sw[c & 1] + gq * 64 * 40, f * 16, lane, 40);
      acc[f] = wmma_bf16(a, b, acc[f]);
    }
  }

  // ---- spill gate pre-activations (+biases) to LDS ----
#pragma unroll
  for (int f = 0; f < 4; ++f) {
#pragma unroll
    for (int r = 0; r < 8; ++r) {
      int m  = wm * 16 + r + ((lane >> 4) << 3);
      int ch = f * 16 + (lane & 15);
      int n  = gq * H_ + jblk + ch;
      sg[gq][m][ch] = acc[f][r] + bih[n] + bhh[n];
    }
  }
  __syncthreads();

  // ---- fused gates: c' = sig(f)*c + sig(i)*tanh(g); h' = sig(o)*tanh(c') ----
#pragma unroll
  for (int i = 0; i < 8; ++i) {
    int idx = i * 256 + tid;               // 32*64 cells
    int m = idx >> 6, ch = idx & 63;
    int b = mblk + m, j = jblk + ch;
    float gi = sg[0][m][ch], gf = sg[1][m][ch];
    float gg = sg[2][m][ch], go = sg[3][m][ch];
    float c  = sigmoidf_(gf) * c_in[(size_t)b * H_ + j] +
               sigmoidf_(gi) * tanhf(gg);
    float h  = sigmoidf_(go) * tanhf(c);
    c_out[(size_t)b * H_ + j] = c;
    h_out[(size_t)b * H_ + j] = f2bf(h);
    if (hs_out) hs_out[(size_t)b * hs_stride + j] = h;
  }
}

// ---------------------------------------------------------------------------
// Generic WMMA GEMM: C[MxN] = A(bf16, MxK) @ W(bf16, NxK)^T (+bias)(+tanh)
// Block tile 64x128; 8 waves as 2(M) x 4(N); wave tile 32x32.
// Double-buffered LDS with async staging. Grid: (N/128, M/64). K % 32 == 0.
// ---------------------------------------------------------------------------
__global__ __launch_bounds__(256) void wmma_gemm_kernel(
    const unsigned short* __restrict__ A, int lda,
    const unsigned short* __restrict__ W, int ldw,
    const float* __restrict__ bias,
    float* __restrict__ C, int ldc, int K, int act)
{
  __shared__ unsigned short sa[2][64 * 40];
  __shared__ unsigned short sw[2][128 * 40];
  const int tid = threadIdx.x, lane = tid & 31, wave = tid >> 5;
  const int wm = wave >> 2, wn = wave & 3;
  const int mblk = blockIdx.y * 64, nblk = blockIdx.x * 128;
  const int row = tid >> 2, kc = (tid & 3) << 3;

  auto stage = [&](int k0, int buf) {
    stage_b128(A + (size_t)(mblk + row) * lda + k0 + kc,
               &sa[buf][row * 40 + kc]);
#pragma unroll
    for (int i = 0; i < 2; ++i) {
      int r = i * 64 + row;
      stage_b128(W + (size_t)(nblk + r) * ldw + k0 + kc,
                 &sw[buf][r * 40 + kc]);
    }
  };

  floatx8 acc[4];
#pragma unroll
  for (int f = 0; f < 4; ++f)
#pragma unroll
    for (int r = 0; r < 8; ++r) acc[f][r] = 0.0f;

  const int nchunks = K / 32;
  stage(0, 0);
  for (int c = 0; c < nchunks; ++c) {
    ASYNC_WAIT();
    __syncthreads();
    if (c + 1 < nchunks) stage((c + 1) * 32, (c + 1) & 1);
    bf16x16 a0 = frag_a(sa[c & 1], wm * 32,      lane, 40);
    bf16x16 a1 = frag_a(sa[c & 1], wm * 32 + 16, lane, 40);
    bf16x16 b0 = frag_b(sw[c & 1], wn * 32,      lane, 40);
    bf16x16 b1 = frag_b(sw[c & 1], wn * 32 + 16, lane, 40);
    acc[0] = wmma_bf16(a0, b0, acc[0]);
    acc[1] = wmma_bf16(a0, b1, acc[1]);
    acc[2] = wmma_bf16(a1, b0, acc[2]);
    acc[3] = wmma_bf16(a1, b1, acc[3]);
  }

#pragma unroll
  for (int f = 0; f < 4; ++f) {
    int mf = (f >> 1) * 16, nf = (f & 1) * 16;
#pragma unroll
    for (int r = 0; r < 8; ++r) {
      int m = mblk + wm * 32 + mf + r + ((lane >> 4) << 3);
      int n = nblk + wn * 32 + nf + (lane & 15);
      float v = acc[f][r];
      if (bias) v += bias[n];
      if (act) v = tanhf(v);
      C[(size_t)m * ldc + n] = v;
    }
  }
}

// ---------------------------------------------------------------------------
// Small support kernels
// ---------------------------------------------------------------------------
__global__ void fill_zero_kernel(float* p, int n) {
  int i = blockIdx.x * 256 + threadIdx.x;
  if (i < n) p[i] = 0.0f;
}

// out[r*out_stride+c] = c<cols ? bf16(in[r*in_stride+c]) : 0
__global__ void conv_bf16_kernel(unsigned short* out, const float* in,
                                 int rows, int cols, int in_stride, int out_stride) {
  int i = blockIdx.x * 256 + threadIdx.x;
  if (i >= rows * out_stride) return;
  int r = i / out_stride, c = i % out_stride;
  out[i] = (c < cols) ? f2bf(in[(size_t)r * in_stride + c]) : (unsigned short)0;
}

// x (B,240,300) f32 -> xb (240,256,320) bf16, zero-padded K
__global__ void prep_x_kernel(unsigned short* xb, const float* x) {
  int i = blockIdx.x * 256 + threadIdx.x;
  if (i >= TTOT_ * B_ * KP_) return;
  int k = i % KP_;
  int r = i / KP_;
  int t = r >> 8, b = r & 255;
  xb[i] = (k < IN_) ? f2bf(x[((size_t)b * TTOT_ + t) * IN_ + k])
                    : (unsigned short)0;
}

__global__ void copy2d_kernel(float* dst, int dstride, const float* src,
                              int sstride, int cols) {
  int i = blockIdx.x * 256 + threadIdx.x;
  int r = i / cols, c = i % cols;
  dst[(size_t)r * dstride + c] = src[(size_t)r * sstride + c];
}

// scores[b,t] = dot(title_hs[b,t,:], v[b,:]) + bil_b ; one wave per (b,t)
__global__ __launch_bounds__(256) void scores_kernel(
    const float* __restrict__ ths, const float* __restrict__ v,
    const float* __restrict__ bil_b, float* __restrict__ scores) {
  int w = blockIdx.x * 8 + (threadIdx.x >> 5);
  int lane = threadIdx.x & 31;
  int b = w / TLEN_, t = w % TLEN_;
  const float* a  = ths + ((size_t)b * TLEN_ + t) * (2 * H_);
  const float* vv = v + (size_t)b * (2 * H_);
  float s = 0.0f;
  for (int e = lane; e < 2 * H_; e += 32) s += a[e] * vv[e];
#pragma unroll
  for (int m = 16; m > 0; m >>= 1) s += __shfl_xor(s, m, 32);
  if (lane == 0) scores[w] = s + bil_b[0];
}

__global__ void softmax_t_kernel(const float* scores, float* probs) {
  __shared__ float s[TLEN_];
  __shared__ float ssum;
  int b = blockIdx.x, t = threadIdx.x;
  if (t < TLEN_) s[t] = scores[b * TLEN_ + t];
  __syncthreads();
  if (t == 0) {
    float m = -1e30f;
    for (int i = 0; i < TLEN_; ++i) m = fmaxf(m, s[i]);
    float su = 0.0f;
    for (int i = 0; i < TLEN_; ++i) { s[i] = __expf(s[i] - m); su += s[i]; }
    ssum = su;
  }
  __syncthreads();
  if (t < TLEN_) probs[b * TLEN_ + t] = s[t] / ssum;
}

// cat1[b, 0:1024] = sum_t title_hs[b,t,:] * probs[b,t]
__global__ void context_kernel(const float* __restrict__ ths,
                               const float* __restrict__ probs,
                               float* __restrict__ cat1) {
  int i = blockIdx.x * 256 + threadIdx.x;   // B_*1024
  int b = i >> 10, e = i & 1023;
  const float* base = ths + (size_t)b * TLEN_ * (2 * H_) + e;
  const float* p = probs + b * TLEN_;
  float s = 0.0f;
  for (int t = 0; t < TLEN_; ++t) s += base[(size_t)t * (2 * H_)] * p[t];
  cat1[(size_t)b * 2048 + e] = s;   // 2048 stride, first half
}

// logits = ch @ cls_W^T + cls_b ; softmax over 4
__global__ __launch_bounds__(128) void cls_kernel(
    const float* __restrict__ ch, const float* __restrict__ W,
    const float* __restrict__ bias, float* __restrict__ out) {
  int b = blockIdx.x, tid = threadIdx.x;
  float a[NC_] = {0.f, 0.f, 0.f, 0.f};
  for (int k = tid; k < CH_; k += 128) {
    float v = ch[(size_t)b * CH_ + k];
#pragma unroll
    for (int c = 0; c < NC_; ++c) a[c] += v * W[c * CH_ + k];
  }
  __shared__ float red[NC_][128];
#pragma unroll
  for (int c = 0; c < NC_; ++c) red[c][tid] = a[c];
  __syncthreads();
  for (int s = 64; s > 0; s >>= 1) {
    if (tid < s)
#pragma unroll
      for (int c = 0; c < NC_; ++c) red[c][tid] += red[c][tid + s];
    __syncthreads();
  }
  if (tid == 0) {
    float l[NC_], m = -1e30f, su = 0.0f;
#pragma unroll
    for (int c = 0; c < NC_; ++c) { l[c] = red[c][0] + bias[c]; m = fmaxf(m, l[c]); }
#pragma unroll
    for (int c = 0; c < NC_; ++c) { l[c] = __expf(l[c] - m); su += l[c]; }
#pragma unroll
    for (int c = 0; c < NC_; ++c) out[b * NC_ + c] = l[c] / su;
  }
}

// ---------------------------------------------------------------------------
// Host orchestration
// ---------------------------------------------------------------------------
extern "C" void kernel_launch(void* const* d_in, const int* in_sizes, int n_in,
                              void* d_out, int out_size, void* d_ws, size_t ws_size,
                              hipStream_t stream) {
  (void)in_sizes; (void)n_in; (void)out_size; (void)ws_size;
  const float* x         = (const float*)d_in[0];
  const float* title_Wih = (const float*)d_in[1];
  const float* title_Whh = (const float*)d_in[2];
  const float* title_bih = (const float*)d_in[3];
  const float* title_bhh = (const float*)d_in[4];
  const float* body_Wih  = (const float*)d_in[5];
  const float* body_Whh  = (const float*)d_in[6];
  const float* body_bih  = (const float*)d_in[7];
  const float* body_bhh  = (const float*)d_in[8];
  const float* bil_W     = (const float*)d_in[9];
  const float* bil_b     = (const float*)d_in[10];
  const float* attn_W    = (const float*)d_in[11];
  const float* attn_b    = (const float*)d_in[12];
  const float* dec_W     = (const float*)d_in[13];
  const float* dec_b     = (const float*)d_in[14];
  const float* cls_W     = (const float*)d_in[15];
  const float* cls_b     = (const float*)d_in[16];
  float* out = (float*)d_out;

  // ---- workspace bump allocator ----
  char* p = (char*)d_ws;
  auto alloc = [&](size_t bytes) -> void* {
    void* r = (void*)p;
    p += (bytes + 255) & ~(size_t)255;
    return r;
  };
  unsigned short* xb     = (unsigned short*)alloc((size_t)TTOT_ * B_ * KP_ * 2);
  unsigned short* WihT   = (unsigned short*)alloc((size_t)2 * G_ * KP_ * 2);
  unsigned short* WihB   = (unsigned short*)alloc((size_t)2 * G_ * KP_ * 2);
  unsigned short* WhhT   = (unsigned short*)alloc((size_t)2 * G_ * H_ * 2);
  unsigned short* WhhB   = (unsigned short*)alloc((size_t)2 * G_ * H_ * 2);
  unsigned short* bilWb  = (unsigned short*)alloc((size_t)1024 * 1024 * 2);
  unsigned short* attnWb = (unsigned short*)alloc((size_t)1024 * 2048 * 2);
  unsigned short* decWb  = (unsigned short*)alloc((size_t)512 * 2048 * 2);
  float* title_hs = (float*)alloc((size_t)B_ * TLEN_ * 1024 * 4);
  unsigned short* hst = (unsigned short*)alloc((size_t)4 * B_ * H_ * 2); // h states (bf16)
  float* cst          = (float*)alloc((size_t)4 * B_ * H_ * 4);          // c states (f32)
  float* cat1     = (float*)alloc((size_t)B_ * 2048 * 4);
  float* cat2     = (float*)alloc((size_t)B_ * 2048 * 4);
  float* vbuf     = (float*)alloc((size_t)B_ * 1024 * 4);
  float* scores   = (float*)alloc((size_t)B_ * TLEN_ * 4);
  float* probs    = (float*)alloc((size_t)B_ * TLEN_ * 4);
  float* chid     = (float*)alloc((size_t)B_ * CH_ * 4);
  unsigned short* blbf   = (unsigned short*)alloc((size_t)B_ * 1024 * 2);
  unsigned short* cat1bf = (unsigned short*)alloc((size_t)B_ * 2048 * 2);
  unsigned short* cat2bf = (unsigned short*)alloc((size_t)B_ * 2048 * 2);

  unsigned short* h0a = hst + 0 * B_ * H_;  unsigned short* h0b = hst + 1 * B_ * H_;
  unsigned short* h1a = hst + 2 * B_ * H_;  unsigned short* h1b = hst + 3 * B_ * H_;
  float* c0a = cst + 0 * B_ * H_;  float* c0b = cst + 1 * B_ * H_;
  float* c1a = cst + 2 * B_ * H_;  float* c1b = cst + 3 * B_ * H_;

  // ---- conversions / init ----
  prep_x_kernel<<<(TTOT_ * B_ * KP_ + 255) / 256, 256, 0, stream>>>(xb, x);
  conv_bf16_kernel<<<(2 * G_ * KP_ + 255) / 256, 256, 0, stream>>>(WihT, title_Wih, 2 * G_, IN_, IN_, KP_);
  conv_bf16_kernel<<<(2 * G_ * KP_ + 255) / 256, 256, 0, stream>>>(WihB, body_Wih, 2 * G_, IN_, IN_, KP_);
  conv_bf16_kernel<<<(2 * G_ * H_ + 255) / 256, 256, 0, stream>>>(WhhT, title_Whh, 2 * G_, H_, H_, H_);
  conv_bf16_kernel<<<(2 * G_ * H_ + 255) / 256, 256, 0, stream>>>(WhhB, body_Whh, 2 * G_, H_, H_, H_);
  conv_bf16_kernel<<<(1024 * 1024 + 255) / 256, 256, 0, stream>>>(bilWb, bil_W, 1024, 1024, 1024, 1024);
  conv_bf16_kernel<<<(1024 * 2048 + 255) / 256, 256, 0, stream>>>(attnWb, attn_W, 1024, 2048, 2048, 2048);
  conv_bf16_kernel<<<(512 * 2048 + 255) / 256, 256, 0, stream>>>(decWb, dec_W, 512, 2048, 2048, 2048);
  fill_zero_kernel<<<(4 * B_ * H_ / 2 + 255) / 256, 256, 0, stream>>>((float*)hst, 4 * B_ * H_ / 2);
  fill_zero_kernel<<<(4 * B_ * H_ + 255) / 256, 256, 0, stream>>>(cst, 4 * B_ * H_);

  dim3 stepGrid(G_ / 4 / 64, B_ / 32);   // (8, 8)

  // ---- dir 0: title forward then body forward (state chains through) ----
  {
    unsigned short *hc = h0a, *hn = h0b;
    float *cc = c0a, *cn = c0b;
    for (int t = 0; t < TLEN_; ++t) {
      lstm_step_kernel<<<stepGrid, 256, 0, stream>>>(
          xb + (size_t)t * B_ * KP_, WihT, WhhT, title_bih, title_bhh,
          hc, cc, hn, cn, title_hs + (size_t)t * 1024, TLEN_ * 1024);
      unsigned short* th = hc; hc = hn; hn = th;
      float* tc = cc; cc = cn; cn = tc;
    }
    for (int s = 0; s < BLEN_; ++s) {
      int t = TLEN_ + s;
      float* hs = (s == BLEN_ - 1) ? (cat1 + 1024) : nullptr;  // body_last[:, :512]
      lstm_step_kernel<<<stepGrid, 256, 0, stream>>>(
          xb + (size_t)t * B_ * KP_, WihB, WhhB, body_bih, body_bhh,
          hc, cc, hn, cn, hs, 2048);
      unsigned short* th = hc; hc = hn; hn = th;
      float* tc = cc; cc = cn; cn = tc;
    }
  }

  // ---- dir 1: title backward (t=39..0) then ONE body-backward step (t=239) ----
  // (body_out[:, -1, :] backward half == state after first reverse step)
  {
    unsigned short *hc = h1a, *hn = h1b;
    float *cc = c1a, *cn = c1b;
    for (int s = 0; s < TLEN_; ++s) {
      int t = TLEN_ - 1 - s;
      lstm_step_kernel<<<stepGrid, 256, 0, stream>>>(
          xb + (size_t)t * B_ * KP_, WihT + (size_t)G_ * KP_, WhhT + (size_t)G_ * H_,
          title_bih + G_, title_bhh + G_,
          hc, cc, hn, cn, title_hs + (size_t)t * 1024 + H_, TLEN_ * 1024);
      unsigned short* th = hc; hc = hn; hn = th;
      float* tc = cc; cc = cn; cn = tc;
    }
    lstm_step_kernel<<<stepGrid, 256, 0, stream>>>(
        xb + (size_t)(TTOT_ - 1) * B_ * KP_, WihB + (size_t)G_ * KP_, WhhB + (size_t)G_ * H_,
        body_bih + G_, body_bhh + G_,
        hc, cc, hn, cn, cat1 + 1536, 2048);                    // body_last[:, 512:]
  }

  // ---- bilinear attention: v = body_last @ bilW^T ----
  conv_bf16_kernel<<<(B_ * 1024 + 255) / 256, 256, 0, stream>>>(blbf, cat1 + 1024, B_, 1024, 2048, 1024);
  wmma_gemm_kernel<<<dim3(1024 / 128, B_ / 64), 256, 0, stream>>>(
      blbf, 1024, bilWb, 1024, nullptr, vbuf, 1024, 1024, 0);
  scores_kernel<<<(B_ * TLEN_) / 8, 256, 0, stream>>>(title_hs, vbuf, bil_b, scores);
  softmax_t_kernel<<<B_, 64, 0, stream>>>(scores, probs);
  context_kernel<<<(B_ * 1024) / 256, 256, 0, stream>>>(title_hs, probs, cat1);

  // ---- attn = tanh(cat1 @ attn_W^T + attn_b) -> cat2[:, 1024:] ----
  conv_bf16_kernel<<<(B_ * 2048 + 255) / 256, 256, 0, stream>>>(cat1bf, cat1, B_, 2048, 2048, 2048);
  wmma_gemm_kernel<<<dim3(1024 / 128, B_ / 64), 256, 0, stream>>>(
      cat1bf, 2048, attnWb, 2048, attn_b, cat2 + 1024, 2048, 2048, 1);
  // cat2[:, :1024] = title_out[:, -1, :]
  copy2d_kernel<<<(B_ * 1024) / 256, 256, 0, stream>>>(
      cat2, 2048, title_hs + (size_t)(TLEN_ - 1) * 1024, TLEN_ * 1024, 1024);

  // ---- class_hidden = cat2 @ dec_W^T + dec_b ----
  conv_bf16_kernel<<<(B_ * 2048 + 255) / 256, 256, 0, stream>>>(cat2bf, cat2, B_, 2048, 2048, 2048);
  wmma_gemm_kernel<<<dim3(CH_ / 128, B_ / 64), 256, 0, stream>>>(
      cat2bf, 2048, decWb, 2048, dec_b, chid, CH_, 2048, 0);

  // ---- classifier + softmax ----
  cls_kernel<<<B_, 128, 0, stream>>>(chid, cls_W, cls_b, out);
}
